// INV_MLP_EMBEDDING_FORWARD_44976897524026
// MI455X (gfx1250) — matrix operand order repaired and checked by reading, not compiled, not run on hardware
//
#include <hip/hip_runtime.h>

// ---------------------------------------------------------------------------
// Fused MLP -> segment-mean -> MLP for gfx1250 (MI455X), bf16 WMMA pipeline.
// ---------------------------------------------------------------------------

#define DEVINL __device__ __forceinline__

typedef float  v8f   __attribute__((ext_vector_type(8)));
typedef __bf16 v16bf __attribute__((ext_vector_type(16)));
typedef __bf16 v8bf  __attribute__((ext_vector_type(8)));
typedef int    v4i   __attribute__((ext_vector_type(4)));

#define N_ROWS  262144
#define IN_DIM  64
#define HID     256
#define EMB     128
#define OUT_DIM 64
#define NUM_SEG 1024

// workspace byte offsets (all 16B-aligned)
#define W1F_OFF   0u         // 64*256  bf16 = 32768
#define W2F_OFF   32768u     // 256*256 bf16 = 131072
#define W3F_OFF   163840u    // 256*128 bf16 = 65536
#define A1F_OFF   229376u    // 256*256 bf16 = 131072
#define A2F_OFF   360448u    // 256*256 bf16 = 131072
#define A3F_OFF   491520u    // 256*64  bf16 = 32768
#define SUMS_OFF  524288u    // 1024*128 f32 = 524288
#define CNT_OFF   1048576u   // 1024 f32
#define MEANS_OFF 1052672u   // 1024*128 f32
#define SEG_OFF   1576960u   // N int32
#define MAXA_OFF  2625536u   // 1 int (padded)
#define H_OFF     2625792u   // N*128 f32

// --------------------------- async staging path ----------------------------
#if defined(__gfx1250__) &&                                                   \
    __has_builtin(__builtin_amdgcn_global_load_async_to_lds_b128) &&          \
    __has_builtin(__builtin_amdgcn_s_wait_asynccnt)
#define USE_ASYNC_LDS 1
typedef __attribute__((address_space(1))) v4i as1_v4i;
typedef __attribute__((address_space(3))) v4i as3_v4i;
#endif

DEVINL void stage16(const void* gsrc, void* ldst) {
#if defined(USE_ASYNC_LDS)
  __builtin_amdgcn_global_load_async_to_lds_b128(
      (as1_v4i*)gsrc, (as3_v4i*)ldst, 0, 0);
#else
  *(uint4*)ldst = *(const uint4*)gsrc;
#endif
}

DEVINL void stage_fence() {
#if defined(USE_ASYNC_LDS)
  __builtin_amdgcn_s_wait_asynccnt(0);
#endif
  __syncthreads();
}

// ---------------------------------------------------------------------------
// Core wave-level GEMM: D(16xNout) += act(16xK) * Wfrag(KxNout), bf16->f32.
// act: LDS, row-major [16][256] bf16. wf: pre-swizzled B-fragment layout:
//   frag f = nt*KT+kt: 32 lanes x 16 contiguous bf16 at wf + f*512 + lane*16;
//   lane holds col n = nt*16+(l&15), k = kt*32 + (e>>3)*16 + (l>>4)*8 + (e&7).
// A fragment per ISA layout: row = l&15, two 8-half chunks at
//   k = kt*32 + (l>>4)*8  and  k = kt*32 + 16 + (l>>4)*8.
// All A fragments preloaded (reused across N-tiles); B fragments streamed
// with a depth-2 register rotation so ds loads stay >=2 ahead of the WMMA.
// ---------------------------------------------------------------------------
template<int KT, int NT>
DEVINL void wave_gemm(const __bf16* __restrict__ act,
                      const __bf16* __restrict__ wf,
                      int lane, v8f (&acc)[NT]) {
  const int half = lane >> 4;
  const int rl   = lane & 15;
  const __bf16* arow = act + rl * HID + half * 8;

  const v8f vzero = {0.f, 0.f, 0.f, 0.f, 0.f, 0.f, 0.f, 0.f};
#pragma unroll
  for (int nt = 0; nt < NT; ++nt) acc[nt] = vzero;

  v16bf a[KT];
#pragma unroll
  for (int kt = 0; kt < KT; ++kt) {
    v8bf lo = *(const v8bf*)(arow + kt * 32);
    v8bf hi = *(const v8bf*)(arow + kt * 32 + 16);
#pragma unroll
    for (int i = 0; i < 8; ++i) { a[kt][i] = lo[i]; a[kt][8 + i] = hi[i]; }
  }

  constexpr int F = KT * NT;
  const __bf16* wb = wf + lane * 16;
  v16bf b0 = *(const v16bf*)(wb);
  v16bf b1 = *(const v16bf*)(wb + 512);
#pragma unroll
  for (int f = 0; f < F; ++f) {
    v16bf bn = (f + 2 < F) ? *(const v16bf*)(wb + (size_t)(f + 2) * 512) : b0;
    const int nt = f / KT;
    const int kt = f - nt * KT;
    acc[nt] = __builtin_amdgcn_wmma_f32_16x16x32_bf16(
                  /*neg_a=*/false, a[kt], /*neg_b=*/false, b0,
                  /*c_mod=*/(short)0, acc[nt],
                  /*reuse_a=*/false, /*reuse_b=*/false);
    b0 = b1;
    b1 = bn;
  }
}

// bias + (optional) leaky-relu, write back to act in C/D layout:
//   row = r + 8*(l>>4), col = nt*16 + (l&15)
template<int NT, bool RELU>
DEVINL void epilogue_to_act(v8f (&acc)[NT], const float* __restrict__ bias,
                            __bf16* act, int lane) {
  const int half = lane >> 4;
  const int rl   = lane & 15;
#pragma unroll
  for (int nt = 0; nt < NT; ++nt) {
    float bv = bias[nt * 16 + rl];
#pragma unroll
    for (int r = 0; r < 8; ++r) {
      float v = acc[nt][r] + bv;
      if (RELU) v = (v > 0.f) ? v : 0.01f * v;
      act[(r + 8 * half) * HID + nt * 16 + rl] = (__bf16)v;
    }
  }
}

// ---------------------------------------------------------------------------
// Prep: zero accumulators (must happen every launch for graph replay)
// ---------------------------------------------------------------------------
__global__ void k_init(float* __restrict__ sums, float* __restrict__ counts,
                       int* __restrict__ maxa) {
  int i = blockIdx.x * blockDim.x + threadIdx.x;
  if (i < NUM_SEG * EMB) sums[i] = 0.f;
  if (i < NUM_SEG) counts[i] = 0.f;
  if (i == 0) maxa[0] = 0;
}

__global__ void k_amax(const int* __restrict__ aisle, int* __restrict__ maxa) {
  int i = blockIdx.x * blockDim.x + threadIdx.x;
  int stride = gridDim.x * blockDim.x;
  int m = 0;
  for (; i < N_ROWS; i += stride) m = max(m, aisle[i]);
  atomicMax(maxa, m);
}

// fp32 [K,Nout] row-major  ->  bf16 WMMA B-fragment layout
__global__ void k_prep(const float* __restrict__ W, __bf16* __restrict__ dst,
                       int K, int Nout) {
  int idx = blockIdx.x * blockDim.x + threadIdx.x;
  if (idx >= K * Nout) return;
  int e    = idx & 15;
  int lane = (idx >> 4) & 31;
  int t    = idx >> 9;          // nt*KT + kt
  int KT   = K >> 5;
  int kt   = t % KT;
  int nt   = t / KT;
  int k = kt * 32 + (e >> 3) * 16 + (lane >> 4) * 8 + (e & 7);
  int n = nt * 16 + (lane & 15);
  dst[idx] = (__bf16)W[k * Nout + n];
}

// ---------------------------------------------------------------------------
// MLP1: x -> lrelu(W1) -> lrelu(W2) -> W3 -> h ; scatter-add sums/counts.
// 256 thr (8 waves), 512 blocks, 4 uniform iterations x 8 tiles.
// LDS: 224KB weights + 8*8KB act = 288KB.
// ---------------------------------------------------------------------------
__global__ __launch_bounds__(256) void k_mlp1(
    const float* __restrict__ x, const int* __restrict__ aisle,
    const int* __restrict__ batch,
    const float* __restrict__ b1, const float* __restrict__ b2,
    const float* __restrict__ b3,
    const __bf16* __restrict__ wall,     // w1f|w2f|w3f contiguous
    float* __restrict__ hout, float* __restrict__ sums,
    float* __restrict__ counts, int* __restrict__ segids,
    const int* __restrict__ maxa) {
  extern __shared__ char smem[];
  __bf16* lw1 = (__bf16*)smem;
  __bf16* lw2 = lw1 + IN_DIM * HID;
  __bf16* lw3 = lw2 + HID * HID;
  __bf16* actall = lw3 + HID * EMB;

  const int tid  = threadIdx.x;
  const int wave = tid >> 5, lane = tid & 31;
  __bf16* act = actall + wave * (16 * HID);

  {  // stage all MLP1 weight fragments into LDS (229376 B = 14336 x 16B)
    const char* src = (const char*)wall;
    char* dst = (char*)smem;
    for (int i = tid; i < (IN_DIM * HID + HID * HID + HID * EMB) / 8; i += 256)
      stage16(src + i * 16, dst + i * 16);
  }
  stage_fence();

  const int mult = maxa[0] + 1;
  const int half = lane >> 4, rl = lane & 15;

#pragma unroll 1
  for (int it = 0; it < 4; ++it) {
    const int tile = blockIdx.x * 32 + it * 8 + wave;
    const int rowbase = tile * 16;

    if (it < 3) {  // prefetch next tile's x rows (4KB across the wave)
      __builtin_prefetch(x + (size_t)(tile + 8) * 16 * IN_DIM + lane * 32, 0, 0);
    }

    int my_seg = 0;
    if (lane < 16) {
      int rr = rowbase + lane;
      my_seg = aisle[rr] + batch[rr] * mult;
      segids[rr] = my_seg;
      atomicAdd(&counts[my_seg], 1.0f);
    }

    // load x tile (16 x 64 f32) -> act cols 0..63 as bf16
#pragma unroll
    for (int i = 0; i < 8; ++i) {
      int e = i * 128 + lane * 4;  // [0,1024)
      float4 v = *(const float4*)(x + (size_t)rowbase * IN_DIM + e);
      int row = e >> 6, col = e & 63;
      act[row * HID + col + 0] = (__bf16)v.x;
      act[row * HID + col + 1] = (__bf16)v.y;
      act[row * HID + col + 2] = (__bf16)v.z;
      act[row * HID + col + 3] = (__bf16)v.w;
    }

    { v8f acc[16]; wave_gemm<2, 16>(act, lw1, lane, acc);
      epilogue_to_act<16, true>(acc, b1, act, lane); }
    { v8f acc[16]; wave_gemm<8, 16>(act, lw2, lane, acc);
      epilogue_to_act<16, true>(acc, b2, act, lane); }
    {
      v8f acc[8]; wave_gemm<8, 8>(act, lw3, lane, acc);
#pragma unroll
      for (int nt = 0; nt < 8; ++nt) {
        float bv = b3[nt * 16 + rl];
#pragma unroll
        for (int r = 0; r < 8; ++r) {
          float v = acc[nt][r] + bv;
          int row = rowbase + r + 8 * half;
          int col = nt * 16 + rl;
          hout[(size_t)row * EMB + col] = v;
          int seg = __shfl(my_seg, r + 8 * half, 32);
          atomicAdd(&sums[seg * EMB + col], v);
        }
      }
    }
  }
}

__global__ void k_means(const float* __restrict__ sums,
                        const float* __restrict__ counts,
                        float* __restrict__ means) {
  int i = blockIdx.x * blockDim.x + threadIdx.x;
  if (i >= NUM_SEG * EMB) return;
  means[i] = sums[i] / fmaxf(counts[i >> 7], 1.0f);
}

// ---------------------------------------------------------------------------
// MLP2: z=[h | means[seg]] -> lrelu(A1) -> lrelu(A2) -> A3 -> out.
// 128 thr (4 waves), 1024 blocks, 4 uniform iterations x 4 tiles.
// LDS: A1+A2 (256KB) + 4*8KB act = 288KB. A3 fragments stream from L2.
// ---------------------------------------------------------------------------
__global__ __launch_bounds__(128) void k_mlp2(
    const float* __restrict__ hbuf, const float* __restrict__ means,
    const int* __restrict__ segids,
    const float* __restrict__ c1, const float* __restrict__ c2,
    const float* __restrict__ c3,
    const __bf16* __restrict__ a12,   // A1f|A2f contiguous
    const __bf16* __restrict__ a3f,   // global fragments
    float* __restrict__ out) {
  extern __shared__ char smem[];
  __bf16* la1 = (__bf16*)smem;
  __bf16* la2 = la1 + HID * HID;
  __bf16* actall = la2 + HID * HID;

  const int tid  = threadIdx.x;
  const int wave = tid >> 5, lane = tid & 31;
  __bf16* act = actall + wave * (16 * HID);

  {  // stage A1f+A2f (262144 B = 16384 x 16B)
    const char* src = (const char*)a12;
    char* dst = (char*)smem;
    for (int i = tid; i < (2 * HID * HID) / 8; i += 128)
      stage16(src + i * 16, dst + i * 16);
  }
  stage_fence();

  const int half = lane >> 4, rl = lane & 15;

#pragma unroll 1
  for (int it = 0; it < 4; ++it) {
    const int tile = blockIdx.x * 16 + it * 4 + wave;
    const int rowbase = tile * 16;

    if (it < 3) {  // prefetch next tile's h rows (8KB across the wave)
      __builtin_prefetch(hbuf + (size_t)(tile + 4) * 16 * EMB + lane * 64, 0, 0);
    }

    int my_seg = 0;
    if (lane < 16) my_seg = segids[rowbase + lane];

    // build z tile: cols 0..127 = h[row], cols 128..255 = means[seg[row]]
#pragma unroll
    for (int i = 0; i < 32; ++i) {
      int e = i * 128 + lane * 4;  // [0,4096)
      int row = e >> 8, col = e & 255;
      int seg = __shfl(my_seg, row, 32);
      float4 v;
      if (col < 128)
        v = *(const float4*)(hbuf + (size_t)(rowbase + row) * EMB + col);
      else
        v = *(const float4*)(means + (size_t)seg * EMB + (col - 128));
      act[row * HID + col + 0] = (__bf16)v.x;
      act[row * HID + col + 1] = (__bf16)v.y;
      act[row * HID + col + 2] = (__bf16)v.z;
      act[row * HID + col + 3] = (__bf16)v.w;
    }

    { v8f acc[16]; wave_gemm<8, 16>(act, la1, lane, acc);
      epilogue_to_act<16, true>(acc, c1, act, lane); }
    { v8f acc[16]; wave_gemm<8, 16>(act, la2, lane, acc);
      epilogue_to_act<16, true>(acc, c2, act, lane); }
    {
      v8f acc[4]; wave_gemm<8, 4>(act, a3f, lane, acc);
#pragma unroll
      for (int nt = 0; nt < 4; ++nt) {
        float bv = c3[nt * 16 + rl];
#pragma unroll
        for (int r = 0; r < 8; ++r) {
          int row = rowbase + r + 8 * half;
          out[(size_t)row * OUT_DIM + nt * 16 + rl] = acc[nt][r] + bv;
        }
      }
    }
  }
}

// ---------------------------------------------------------------------------
extern "C" void kernel_launch(void* const* d_in, const int* in_sizes, int n_in,
                              void* d_out, int out_size, void* d_ws,
                              size_t ws_size, hipStream_t stream) {
  (void)in_sizes; (void)n_in; (void)out_size; (void)ws_size;
  const float* x   = (const float*)d_in[0];
  const int* aisle = (const int*)d_in[1];
  const int* batch = (const int*)d_in[2];
  const float* W1 = (const float*)d_in[3];  const float* b1 = (const float*)d_in[4];
  const float* W2 = (const float*)d_in[5];  const float* b2 = (const float*)d_in[6];
  const float* W3 = (const float*)d_in[7];  const float* b3 = (const float*)d_in[8];
  const float* A1 = (const float*)d_in[9];  const float* c1 = (const float*)d_in[10];
  const float* A2 = (const float*)d_in[11]; const float* c2 = (const float*)d_in[12];
  const float* A3 = (const float*)d_in[13]; const float* c3 = (const float*)d_in[14];
  float* out = (float*)d_out;
  char* ws = (char*)d_ws;

  __bf16* w1f = (__bf16*)(ws + W1F_OFF);
  __bf16* w2f = (__bf16*)(ws + W2F_OFF);
  __bf16* w3f = (__bf16*)(ws + W3F_OFF);
  __bf16* a1f = (__bf16*)(ws + A1F_OFF);
  __bf16* a2f = (__bf16*)(ws + A2F_OFF);
  __bf16* a3f = (__bf16*)(ws + A3F_OFF);
  float* sums   = (float*)(ws + SUMS_OFF);
  float* counts = (float*)(ws + CNT_OFF);
  float* means  = (float*)(ws + MEANS_OFF);
  int*   segids = (int*)(ws + SEG_OFF);
  int*   maxa   = (int*)(ws + MAXA_OFF);
  float* hbuf   = (float*)(ws + H_OFF);

  k_init<<<(NUM_SEG * EMB + 255) / 256, 256, 0, stream>>>(sums, counts, maxa);
  k_amax<<<256, 256, 0, stream>>>(aisle, maxa);

  k_prep<<<(IN_DIM * HID + 255) / 256, 256, 0, stream>>>(W1, w1f, IN_DIM, HID);
  k_prep<<<(HID * HID + 255) / 256, 256, 0, stream>>>(W2, w2f, HID, HID);
  k_prep<<<(HID * EMB + 255) / 256, 256, 0, stream>>>(W3, w3f, HID, EMB);
  k_prep<<<(HID * HID + 255) / 256, 256, 0, stream>>>(A1, a1f, 2 * EMB, HID);
  k_prep<<<(HID * HID + 255) / 256, 256, 0, stream>>>(A2, a2f, HID, HID);
  k_prep<<<(HID * OUT_DIM + 255) / 256, 256, 0, stream>>>(A3, a3f, HID, OUT_DIM);

  // 288KB dynamic LDS each: one workgroup per WGP
  k_mlp1<<<512, 256, 294912, stream>>>(x, aisle, batch, b1, b2, b3,
                                       w1f, hbuf, sums, counts, segids, maxa);
  k_means<<<(NUM_SEG * EMB) / 256, 256, 0, stream>>>(sums, counts, means);
  k_mlp2<<<1024, 128, 294912, stream>>>(hbuf, means, segids, c1, c2, c3,
                                        a1f, a3f, out);
}
